// LiveNet_4801773437611
// MI455X (gfx1250) — matrix operand order repaired and checked by reading, not compiled, not run on hardware
//
#include <hip/hip_runtime.h>

typedef __attribute__((ext_vector_type(16))) __bf16 v16bf;
typedef __attribute__((ext_vector_type(8)))  float  v8f;
typedef __attribute__((ext_vector_type(4)))  int    v4i;
typedef unsigned short u16;

union FragU {
    v16bf bf;
    uint4 q[2];
};

__device__ __forceinline__ u16 f32_to_bf16_rne(float f) {
    unsigned int u = __float_as_uint(f);
    u += 0x7FFFu + ((u >> 16) & 1u);           // round-to-nearest-even
    return (u16)(u >> 16);
}

__device__ __forceinline__ void split2(float f, u16& h, u16& l) {
    h = f32_to_bf16_rne(f);
    float hf = __uint_as_float((unsigned)h << 16);
    l = f32_to_bf16_rne(f - hf);
}

// ---- CDNA5 async global->LDS copy (ASYNCcnt path), with portable fallback ----
#if __has_builtin(__builtin_amdgcn_global_load_async_to_lds_b128)
#define HAS_ASYNC_LDS 1
#else
#define HAS_ASYNC_LDS 0
#endif

__device__ __forceinline__ void cp16_g2l(const u16* g, u16* l) {
#if HAS_ASYNC_LDS
    // signature (from hipcc diagnostic): (int4 AS1* gsrc, int4 AS3* ldsdst, imm offset, imm cpol)
    __builtin_amdgcn_global_load_async_to_lds_b128(
        (__attribute__((address_space(1))) v4i*)(v4i*)(void*)(u16*)g,
        (__attribute__((address_space(3))) v4i*)(v4i*)(void*)l,
        0, 0);
#else
    *(uint4*)l = *(const uint4*)g;             // global_load_b128 + ds_store_b128
#endif
}

__device__ __forceinline__ void wait_async_all() {
#if HAS_ASYNC_LDS
#if __has_builtin(__builtin_amdgcn_s_wait_asynccnt)
    __builtin_amdgcn_s_wait_asynccnt(0);
#else
    asm volatile("s_wait_asynccnt 0" ::: "memory");
#endif
#endif
}

// ---------------------------------------------------------------------------
// Prep kernel 1: elementwise split fp32 -> (hi, lo) bf16 arrays. n % 4 == 0.
// ---------------------------------------------------------------------------
__global__ __launch_bounds__(256)
void split_f32_bf16(const float* __restrict__ src, u16* __restrict__ hi,
                    u16* __restrict__ lo, long n)
{
    long i = ((long)blockIdx.x * blockDim.x + threadIdx.x) * 4;
    if (i >= n) return;
    float4 f = *(const float4*)(src + i);
    float v[4] = {f.x, f.y, f.z, f.w};
    u16 h[4], l[4];
    #pragma unroll
    for (int j = 0; j < 4; ++j) split2(v[j], h[j], l[j]);
    uint2 ph = {(unsigned)h[0] | ((unsigned)h[1] << 16), (unsigned)h[2] | ((unsigned)h[3] << 16)};
    uint2 pl = {(unsigned)l[0] | ((unsigned)l[1] << 16), (unsigned)l[2] | ((unsigned)l[3] << 16)};
    *(uint2*)(hi + i) = ph;
    *(uint2*)(lo + i) = pl;
}

// ---------------------------------------------------------------------------
// Prep kernel 2: W [K][N] fp32 -> transposed Thi/Tlo [N][K] bf16 (32x32 tiles).
// grid = (N/32, K/32), block = 256.
// ---------------------------------------------------------------------------
__global__ __launch_bounds__(256)
void transpose_split_f32_bf16(const float* __restrict__ W, u16* __restrict__ Thi,
                              u16* __restrict__ Tlo, int K, int N)
{
    __shared__ u16 th[32][33];   // +1 pad: conflict-free transposed access
    __shared__ u16 tl[32][33];
    const int n0 = blockIdx.x * 32;
    const int k0 = blockIdx.y * 32;
    const int r  = threadIdx.x >> 3;        // 0..31
    const int c4 = (threadIdx.x & 7) * 4;   // 0..28

    float4 f = *(const float4*)(W + (long)(k0 + r) * N + n0 + c4);
    float v[4] = {f.x, f.y, f.z, f.w};
    #pragma unroll
    for (int j = 0; j < 4; ++j) {
        u16 h, l;
        split2(v[j], h, l);
        th[c4 + j][r] = h;                  // transposed scatter
        tl[c4 + j][r] = l;
    }
    __syncthreads();
    u16 hh[4], ll[4];
    #pragma unroll
    for (int j = 0; j < 4; ++j) { hh[j] = th[r][c4 + j]; ll[j] = tl[r][c4 + j]; }
    uint2 ph = {(unsigned)hh[0] | ((unsigned)hh[1] << 16), (unsigned)hh[2] | ((unsigned)hh[3] << 16)};
    uint2 pl = {(unsigned)ll[0] | ((unsigned)ll[1] << 16), (unsigned)ll[2] | ((unsigned)ll[3] << 16)};
    long o = (long)(n0 + r) * K + k0 + c4;
    *(uint2*)(Thi + o) = ph;
    *(uint2*)(Tlo + o) = pl;
}

// ---------------------------------------------------------------------------
// GEMM: C = act(A * B + bias), A as hi/lo bf16 [M][K], B as hi/lo bf16 [N][K]
// (transposed). bf16x3 (hi*hi + hi*lo + lo*hi) ~ fp32 accuracy, fp32 accum.
// Block 256 (8 waves, 2x4), tile 64M x 128N, wave tile 32x32.
// Double-buffered LDS filled by async global->LDS copies; 1 barrier / k-step.
// ---------------------------------------------------------------------------
#define LDS_STRIDE 40                       // 32 data + 8 pad halves per row
#define A_HI 0
#define A_LO (64 * LDS_STRIDE)
#define B_HI (2 * 64 * LDS_STRIDE)
#define B_LO (2 * 64 * LDS_STRIDE + 128 * LDS_STRIDE)
#define LDS_TOTAL (2 * 64 * LDS_STRIDE + 2 * 128 * LDS_STRIDE)   // 15360 u16 = 30KB

__global__ __launch_bounds__(256)
void gemm_bf16pair_wmma(const u16* __restrict__ Ahi, const u16* __restrict__ Alo,
                        const u16* __restrict__ Bthi, const u16* __restrict__ Btlo,
                        const float* __restrict__ bias,
                        float* __restrict__ Cf,                       // fp32 out (or null)
                        u16* __restrict__ Chi, u16* __restrict__ Clo, // split out (or null)
                        int M, int N, int K, int relu)
{
    __shared__ u16 lds[2 * LDS_TOTAL];      // double buffer, 60KB

    const int tid    = threadIdx.x;
    const int lane   = tid & 31;
    const int wid    = tid >> 5;
    const int wave_m = wid >> 2;       // 0..1
    const int wave_n = wid & 3;        // 0..3
    const int g      = lane >> 4;
    const int lm     = lane & 15;

    const int tileN = blockIdx.x * 128;
    const int tileM = blockIdx.y * 64;

    // staging roles (pure b128 copies)
    const int arow = tid >> 2;          // 0..63
    const int aseg = tid & 3;           // halves aseg*8
    const int bn   = tid >> 1;          // 0..127
    const int bseg = tid & 1;           // halves bseg*16

    const long aoff = (long)(tileM + arow) * K + aseg * 8;
    const long boff = (long)(tileN + bn) * K + bseg * 16;
    const int  la   = arow * LDS_STRIDE + aseg * 8;   // LDS slot for A piece
    const int  lb   = bn * LDS_STRIDE + bseg * 16;    // LDS slot for B piece

    v8f acc[2][2];
    #pragma unroll
    for (int mi = 0; mi < 2; ++mi)
        #pragma unroll
        for (int ni = 0; ni < 2; ++ni)
            acc[mi][ni] = (v8f){0.f, 0.f, 0.f, 0.f, 0.f, 0.f, 0.f, 0.f};

    // issue tile 0 into buffer 0
    {
        u16* buf = lds;
        cp16_g2l(Ahi + aoff, buf + A_HI + la);
        cp16_g2l(Alo + aoff, buf + A_LO + la);
        cp16_g2l(Bthi + boff,     buf + B_HI + lb);
        cp16_g2l(Bthi + boff + 8, buf + B_HI + lb + 8);
        cp16_g2l(Btlo + boff,     buf + B_LO + lb);
        cp16_g2l(Btlo + boff + 8, buf + B_LO + lb + 8);
    }

    const int nIter = K >> 5;
    for (int it = 0; it < nIter; ++it) {
        const int cur = it & 1;
        u16* buf = lds + cur * LDS_TOTAL;

        wait_async_all();        // this wave's slice of buf[cur] has landed
        __syncthreads();         // everyone's slice landed; prev reads of buf[1-cur] drained

        // prefetch next tile into the other buffer (a full k-step of latency hiding)
        if (it + 1 < nIter) {
            u16* nbuf = lds + (cur ^ 1) * LDS_TOTAL;
            const long ao = aoff + (long)(it + 1) * 32;
            const long bo = boff + (long)(it + 1) * 32;
            cp16_g2l(Ahi + ao, nbuf + A_HI + la);
            cp16_g2l(Alo + ao, nbuf + A_LO + la);
            cp16_g2l(Bthi + bo,     nbuf + B_HI + lb);
            cp16_g2l(Bthi + bo + 8, nbuf + B_HI + lb + 8);
            cp16_g2l(Btlo + bo,     nbuf + B_LO + lb);
            cp16_g2l(Btlo + bo + 8, nbuf + B_LO + lb + 8);
        }

        // fragment loads (layouts per CDNA5 ISA 7.12.2)
        FragU a_hi[2], a_lo[2], b_hi[2], b_lo[2];
        #pragma unroll
        for (int mi = 0; mi < 2; ++mi) {
            const int row  = wave_m * 32 + mi * 16 + lm;
            const int base = row * LDS_STRIDE + 8 * g;
            a_hi[mi].q[0] = *(const uint4*)&buf[A_HI + base];
            a_hi[mi].q[1] = *(const uint4*)&buf[A_HI + base + 16];
            a_lo[mi].q[0] = *(const uint4*)&buf[A_LO + base];
            a_lo[mi].q[1] = *(const uint4*)&buf[A_LO + base + 16];
        }
        #pragma unroll
        for (int ni = 0; ni < 2; ++ni) {
            const int n    = wave_n * 32 + ni * 16 + lm;
            const int base = n * LDS_STRIDE + 8 * g;
            b_hi[ni].q[0] = *(const uint4*)&buf[B_HI + base];
            b_hi[ni].q[1] = *(const uint4*)&buf[B_HI + base + 16];
            b_lo[ni].q[0] = *(const uint4*)&buf[B_LO + base];
            b_lo[ni].q[1] = *(const uint4*)&buf[B_LO + base + 16];
        }

        #pragma unroll
        for (int mi = 0; mi < 2; ++mi)
            #pragma unroll
            for (int ni = 0; ni < 2; ++ni) {
                acc[mi][ni] = __builtin_amdgcn_wmma_f32_16x16x32_bf16(
                    false, a_hi[mi].bf, false, b_hi[ni].bf, (short)0, acc[mi][ni], false, false);
                acc[mi][ni] = __builtin_amdgcn_wmma_f32_16x16x32_bf16(
                    false, a_hi[mi].bf, false, b_lo[ni].bf, (short)0, acc[mi][ni], false, false);
                acc[mi][ni] = __builtin_amdgcn_wmma_f32_16x16x32_bf16(
                    false, a_lo[mi].bf, false, b_hi[ni].bf, (short)0, acc[mi][ni], false, false);
            }
    }

    // epilogue: bias (+ReLU). C/D layout: VGPR r -> M = r + 8g, lane -> N.
    #pragma unroll
    for (int mi = 0; mi < 2; ++mi) {
        #pragma unroll
        for (int ni = 0; ni < 2; ++ni) {
            const int col = tileN + wave_n * 32 + ni * 16 + lm;
            const float bv = bias[col];
            #pragma unroll
            for (int r = 0; r < 8; ++r) {
                const int row = tileM + wave_m * 32 + mi * 16 + r + 8 * g;
                float v = acc[mi][ni][r] + bv;
                if (relu) v = fmaxf(v, 0.0f);
                const long o = (long)row * N + col;
                if (Cf) {
                    Cf[o] = v;
                } else {
                    u16 h, l;
                    split2(v, h, l);
                    Chi[o] = h;
                    Clo[o] = l;
                }
            }
        }
    }
}

// ---------------------------------------------------------------------------
extern "C" void kernel_launch(void* const* d_in, const int* in_sizes, int n_in,
                              void* d_out, int out_size, void* d_ws, size_t ws_size,
                              hipStream_t stream) {
    (void)in_sizes; (void)n_in; (void)out_size; (void)ws_size;
    const float* x  = (const float*)d_in[0];   // [4096,1024]
    const float* W1 = (const float*)d_in[1];   // [1024,4096]
    const float* b1 = (const float*)d_in[2];   // [4096]
    const float* W2 = (const float*)d_in[3];   // [4096,1024]
    const float* b2 = (const float*)d_in[4];   // [1024]
    float* Y = (float*)d_out;                  // [4096,1024]

    const size_t MB = 1u << 20;
    char* ws = (char*)d_ws;                    // 112 MB total
    u16* Xhi   = (u16*)(ws + 0 * MB);          //  8 MB  [4096][1024]
    u16* Xlo   = (u16*)(ws + 8 * MB);          //  8 MB
    u16* W1thi = (u16*)(ws + 16 * MB);         //  8 MB  [4096][1024] (transposed)
    u16* W1tlo = (u16*)(ws + 24 * MB);         //  8 MB
    u16* W2thi = (u16*)(ws + 32 * MB);         //  8 MB  [1024][4096] (transposed)
    u16* W2tlo = (u16*)(ws + 40 * MB);         //  8 MB
    u16* Hhi   = (u16*)(ws + 48 * MB);         // 32 MB  [4096][4096]
    u16* Hlo   = (u16*)(ws + 80 * MB);         // 32 MB

    dim3 blk(256);

    // one-time conversions (bandwidth-trivial)
    split_f32_bf16<<<(4096L * 1024 / 4) / 256, blk, 0, stream>>>(x, Xhi, Xlo, 4096L * 1024);
    transpose_split_f32_bf16<<<dim3(4096 / 32, 1024 / 32), blk, 0, stream>>>(W1, W1thi, W1tlo, 1024, 4096);
    transpose_split_f32_bf16<<<dim3(1024 / 32, 4096 / 32), blk, 0, stream>>>(W2, W2thi, W2tlo, 4096, 1024);

    // Layer 1: H = relu(x @ W1 + b1)   M=4096 N=4096 K=1024  (split bf16 out)
    gemm_bf16pair_wmma<<<dim3(4096 / 128, 4096 / 64), blk, 0, stream>>>(
        Xhi, Xlo, W1thi, W1tlo, b1, nullptr, Hhi, Hlo, 4096, 4096, 1024, 1);
    // Layer 2: Y = H @ W2 + b2         M=4096 N=1024 K=4096  (fp32 out)
    gemm_bf16pair_wmma<<<dim3(1024 / 128, 4096 / 64), blk, 0, stream>>>(
        Hhi, Hlo, W2thi, W2tlo, b2, Y, nullptr, nullptr, 4096, 1024, 4096, 0);
}
